// MambaBlock_26903675142267
// MI455X (gfx1250) — matrix operand order repaired and checked by reading, not compiled, not run on hardware
//
#include <hip/hip_runtime.h>
#include <hip/hip_bf16.h>

// ---------------------------------------------------------------------------
// Mamba block for MI455X (gfx1250), wave32 + WMMA bf16 (fp32 accumulate).
//
// GEMMs are guard-free (M%64==0, N%128==0, K%64==0) so EXEC stays all-ones
// around every v_wmma. The K-loop is software-pipelined (ping-pong fragment
// register sets) so each stage's 10x global_load_b128 clause issues while the
// previous stage's 4 WMMAs execute -> partial s_wait_loadcnt, not 0x0.
// ---------------------------------------------------------------------------

typedef __bf16 bf16_t;
typedef bf16_t v16bf __attribute__((ext_vector_type(16)));
typedef bf16_t v8bf  __attribute__((ext_vector_type(8)));
typedef float  v8f   __attribute__((ext_vector_type(8)));

#define D_MODEL 1024
#define D_INNER 2048
#define D_STATE 16
#define D_CONV  4
#define DT_RANK 64
#define N_XDBL  (DT_RANK + 2 * D_STATE)   // 96 valid cols
#define XDBL_LD 128                        // padded row stride / GEMM N
#define BATCH   4
#define SEQ     1024
#define NTOK    (BATCH * SEQ)              // 4096 rows (GEMM M)

__device__ __forceinline__ float siluf(float x) {
    return x / (1.0f + __expf(-x));
}
__device__ __forceinline__ float softplusf(float x) {
    return (x > 20.0f) ? x : log1pf(__expf(x));
}

// ------------------------- fragment loaders --------------------------------
// A fragment (16x32 bf16, M x K). lane<16: M=lane, K = 0..7 & 16..23;
// lane>=16: same row, K = 8..15 & 24..31. Two contiguous 16B loads.
__device__ __forceinline__ v16bf load_a_frag(const bf16_t* __restrict__ A,
                                             int lda, int row, int k0, int half) {
    const bf16_t* p = A + (size_t)row * lda + k0 + half * 8;
    v8bf lo = *(const v8bf*)p;
    v8bf hi = *(const v8bf*)(p + 16);
    v16bf r;
#pragma unroll
    for (int i = 0; i < 8; ++i) { r[i] = lo[i]; r[i + 8] = hi[i]; }
    return r;
}

// B fragment (32x16 bf16, K x N) with B = W^T, W stored [N,K] row-major.
// lane holds column N=lane&15; lane<16 holds K=0..7 & 16..23, lane>=16 K=8..15 & 24..31.
// => two contiguous 16B loads from one row of W.
__device__ __forceinline__ v16bf load_b_frag(const bf16_t* __restrict__ wrow,
                                             int k0, int half) {
    const bf16_t* p = wrow + k0 + half * 8;
    v8bf lo = *(const v8bf*)p;
    v8bf hi = *(const v8bf*)(p + 16);
    v16bf r;
#pragma unroll
    for (int i = 0; i < 8; ++i) { r[i] = lo[i]; r[i + 8] = hi[i]; }
    return r;
}

// One pipeline stage: A fragment + 4 B fragments (one 16x32 k-slice of the
// wave's 16x64 output tile).
struct Stage {
    v16bf a, b0, b1, b2, b3;
};

__device__ __forceinline__ Stage load_stage(const bf16_t* __restrict__ A, int K,
                                            int arow,
                                            const bf16_t* __restrict__ w0,
                                            const bf16_t* __restrict__ w1,
                                            const bf16_t* __restrict__ w2,
                                            const bf16_t* __restrict__ w3,
                                            int k0, int half) {
    Stage s;
    s.a  = load_a_frag(A, K, arow, k0, half);
    s.b0 = load_b_frag(w0, k0, half);
    s.b1 = load_b_frag(w1, k0, half);
    s.b2 = load_b_frag(w2, k0, half);
    s.b3 = load_b_frag(w3, k0, half);
    return s;
}

__device__ __forceinline__ void mma_stage(const Stage& s, v8f acc[4]) {
    acc[0] = __builtin_amdgcn_wmma_f32_16x16x32_bf16(false, s.a, false, s.b0,
                                                     (short)0, acc[0], false, false);
    acc[1] = __builtin_amdgcn_wmma_f32_16x16x32_bf16(false, s.a, false, s.b1,
                                                     (short)0, acc[1], false, false);
    acc[2] = __builtin_amdgcn_wmma_f32_16x16x32_bf16(false, s.a, false, s.b2,
                                                     (short)0, acc[2], false, false);
    acc[3] = __builtin_amdgcn_wmma_f32_16x16x32_bf16(false, s.a, false, s.b3,
                                                     (short)0, acc[3], false, false);
}

// ------------------------------ GEMM ---------------------------------------
// C[M,N] = A[M,K](bf16, lda=K) * W[N,K]^T(bf16).  M%64==0, N%128==0, K%64==0.
// Grid: (N/128, M/64), block 256 = 8 waves (4 along M x 2 along N);
// wave tile 16(M) x 64(N) = 4 accumulators, A fragment reused 4x.
// mode 0: C = acc
// mode 1: C = softplus(acc + bias[n])
// mode 2: C += acc * rowmask[m]        (masked residual accumulate)
__global__ __launch_bounds__(256) void wmma_gemm_kernel(
    const bf16_t* __restrict__ A, const bf16_t* __restrict__ W,
    float* __restrict__ C, int M, int N, int K, int mode,
    const float* __restrict__ bias, const unsigned char* __restrict__ rowmask) {
    const int tid  = threadIdx.x;
    const int lane = tid & 31;
    const int wave = tid >> 5;
    const int ln   = lane & 15;
    const int half = lane >> 4;
    const int wm   = wave & 3;       // 4 waves along M
    const int wn   = wave >> 2;      // 2 waves along N
    const int m0   = blockIdx.y * 64 + wm * 16;
    const int n0b  = blockIdx.x * 128 + wn * 64;

    v8f acc[4];
#pragma unroll
    for (int j = 0; j < 4; ++j)
#pragma unroll
        for (int v = 0; v < 8; ++v) acc[j][v] = 0.0f;

    const bf16_t* w0 = W + (size_t)(n0b + 0 * 16 + ln) * K;
    const bf16_t* w1 = W + (size_t)(n0b + 1 * 16 + ln) * K;
    const bf16_t* w2 = W + (size_t)(n0b + 2 * 16 + ln) * K;
    const bf16_t* w3 = W + (size_t)(n0b + 3 * 16 + ln) * K;
    const int arow = m0 + ln;

    // Software-pipelined over 32-wide k slices, two slices per iteration.
    Stage s0 = load_stage(A, K, arow, w0, w1, w2, w3, 0, half);
    int k0 = 0;
    for (; k0 + 64 < K; k0 += 64) {
        Stage s1 = load_stage(A, K, arow, w0, w1, w2, w3, k0 + 32, half);
        mma_stage(s0, acc);
        s0 = load_stage(A, K, arow, w0, w1, w2, w3, k0 + 64, half);
        mma_stage(s1, acc);
    }
    Stage s1 = load_stage(A, K, arow, w0, w1, w2, w3, k0 + 32, half);
    mma_stage(s0, acc);
    mma_stage(s1, acc);

    // Epilogue: lane (ln,half), vgpr v -> (m = m0 + half*8 + v, n = n0 + ln)
#pragma unroll
    for (int j = 0; j < 4; ++j) {
        const int n = n0b + j * 16 + ln;
#pragma unroll
        for (int v = 0; v < 8; ++v) {
            const int m = m0 + half * 8 + v;
            const size_t idx = (size_t)m * N + n;
            const float val = acc[j][v];
            if (mode == 0) {
                C[idx] = val;
            } else if (mode == 1) {
                C[idx] = softplusf(val + bias[n]);
            } else {
                const float mf = rowmask[m] ? 1.0f : 0.0f;
                C[idx] += val * mf;
            }
        }
    }
}

// ----------------------- elementwise / prep kernels ------------------------
__global__ __launch_bounds__(256) void f32_to_bf16_kernel(
    const float* __restrict__ in, bf16_t* __restrict__ out, size_t n) {
    for (size_t i = (size_t)blockIdx.x * blockDim.x + threadIdx.x; i < n;
         i += (size_t)gridDim.x * blockDim.x)
        out[i] = (bf16_t)in[i];
}

// f32 (rows_src x cols) -> bf16 (rows_dst x cols), zero-padding extra rows
__global__ __launch_bounds__(256) void f32_to_bf16_padrows_kernel(
    const float* __restrict__ in, bf16_t* __restrict__ out,
    int rows_src, int cols, size_t n_out) {
    const size_t i = (size_t)blockIdx.x * blockDim.x + threadIdx.x;
    if (i >= n_out) return;
    const int r = (int)(i / cols);
    const int c = (int)(i % cols);
    out[i] = (r < rows_src) ? (bf16_t)in[(size_t)r * cols + c] : (bf16_t)0.0f;
}

// residual = x*mask -> d_out ; normed = rmsnorm(x)*w -> bf16 (one block/token)
__global__ __launch_bounds__(256) void rms_prep_kernel(
    const float* __restrict__ x, const unsigned char* __restrict__ mask,
    const float* __restrict__ w, float* __restrict__ resid,
    bf16_t* __restrict__ normed_bf) {
    __shared__ float red[256];
    const int tok = blockIdx.x;
    const int tid = threadIdx.x;
    const float* xr = x + (size_t)tok * D_MODEL;
    float s = 0.0f;
    for (int i = tid; i < D_MODEL; i += 256) { float v = xr[i]; s += v * v; }
    red[tid] = s;
    __syncthreads();
    for (int off = 128; off > 0; off >>= 1) {
        if (tid < off) red[tid] += red[tid + off];
        __syncthreads();
    }
    const float rs = rsqrtf(red[0] / (float)D_MODEL + 1e-6f);
    const float mf = mask[tok] ? 1.0f : 0.0f;
    for (int i = tid; i < D_MODEL; i += 256) {
        const float v = xr[i];
        resid[(size_t)tok * D_MODEL + i] = v * mf;
        normed_bf[(size_t)tok * D_MODEL + i] = (bf16_t)(v * rs * w[i]);
    }
}

// segmented causal depthwise conv (taps gated off across invalid steps) + SiLU
__global__ __launch_bounds__(256) void conv_silu_kernel(
    const float* __restrict__ xz, const unsigned char* __restrict__ mask,
    const float* __restrict__ convw, const float* __restrict__ convb,
    float* __restrict__ x_c, bf16_t* __restrict__ x_c_bf, size_t n) {
    const size_t idx = (size_t)blockIdx.x * blockDim.x + threadIdx.x;
    if (idx >= n) return;
    const int d = (int)(idx % D_INNER);
    const size_t bt = idx / D_INNER;           // flat (b,t)
    const int t = (int)(bt % SEQ);
    float acc = convb[d];
    float g = mask[bt] ? 1.0f : 0.0f;          // gate for tap 0
    acc += convw[d * 4 + 3] * xz[bt * (2 * D_INNER) + d] * g;
#pragma unroll
    for (int j = 1; j < D_CONV; ++j) {
        if (t >= j) {
            g *= mask[bt - j] ? 1.0f : 0.0f;   // all of t-j..t must be valid
            acc += convw[d * 4 + (3 - j)] * xz[(bt - j) * (2 * D_INNER) + d] * g;
        }
    }
    const float xc = siluf(acc);
    x_c[idx]    = xc;
    x_c_bf[idx] = (bf16_t)xc;
}

// dt_r = x_dbl[:, :DT_RANK] -> bf16 (strided slice, src stride XDBL_LD)
__global__ __launch_bounds__(256) void dtr_slice_kernel(
    const float* __restrict__ x_dbl, bf16_t* __restrict__ dtr_bf, size_t n) {
    const size_t i = (size_t)blockIdx.x * blockDim.x + threadIdx.x;
    if (i >= n) return;
    const size_t m = i >> 6;       // row
    const int    r = (int)(i & 63);
    dtr_bf[i] = (bf16_t)x_dbl[m * XDBL_LD + r];
}

// SSM scan: lane per (b,d); h[16] in regs; B/C rows staged through LDS.
// Emits ybf = (scan + D*x_c) * silu(z) as bf16 (A operand of out_proj).
__global__ __launch_bounds__(256) void scan_kernel(
    const float* __restrict__ x_dbl, const float* __restrict__ dt,
    const float* __restrict__ x_c, const float* __restrict__ xz,
    const unsigned char* __restrict__ mask, const float* __restrict__ A_log,
    const float* __restrict__ Dco, bf16_t* __restrict__ ybf) {
    __shared__ float sBC[64 * 32];   // 64 timesteps x (B[16] | C[16])
    __shared__ float smask[64];
    const int tid = threadIdx.x;
    const int d = blockIdx.x * 256 + tid;
    const int b = blockIdx.y;

    float As[D_STATE];
#pragma unroll
    for (int s = 0; s < D_STATE; ++s) As[s] = -__expf(A_log[(size_t)d * D_STATE + s]);
    const float Dd = Dco[d];
    float h[D_STATE];
#pragma unroll
    for (int s = 0; s < D_STATE; ++s) h[s] = 0.0f;

    for (int tc = 0; tc < SEQ; tc += 64) {
        for (int i = tid; i < 64 * 32; i += 256) {
            const int tl = i >> 5, c = i & 31;
            sBC[i] = x_dbl[(size_t)(b * SEQ + tc + tl) * XDBL_LD + DT_RANK + c];
        }
        if (tid < 64) smask[tid] = mask[(size_t)b * SEQ + tc + tid] ? 1.0f : 0.0f;
        __syncthreads();

        for (int tl = 0; tl < 64; ++tl) {
            const size_t row = (size_t)b * SEQ + tc + tl;
            const float dtv = dt[row * D_INNER + d];
            const float xcv = x_c[row * D_INNER + d];
            const float zv  = xz[row * (2 * D_INNER) + D_INNER + d];
            const float* Bp = &sBC[tl * 32];
            const float* Cp = Bp + 16;
            float y = 0.0f;
            if (smask[tl] > 0.0f) {            // uniform across block
                const float dBc = dtv * xcv;
#pragma unroll
                for (int s = 0; s < D_STATE; ++s) {
                    const float dA = __expf(dtv * As[s]);
                    h[s] = dA * h[s] + dBc * Bp[s];
                    y += h[s] * Cp[s];
                }
            } else {
#pragma unroll
                for (int s = 0; s < D_STATE; ++s) h[s] = 0.0f;  // state reset
            }
            const float out = (y + Dd * xcv) * siluf(zv);
            ybf[row * D_INNER + d] = (bf16_t)out;
        }
        __syncthreads();
    }
}

// ------------------------------ launcher -----------------------------------
extern "C" void kernel_launch(void* const* d_in, const int* in_sizes, int n_in,
                              void* d_out, int out_size, void* d_ws, size_t ws_size,
                              hipStream_t stream) {
    (void)in_sizes; (void)n_in; (void)out_size; (void)ws_size;
    const float*         x        = (const float*)d_in[0];
    const unsigned char* mask     = (const unsigned char*)d_in[1];  // jnp bool
    const float*         rms_w    = (const float*)d_in[2];
    const float*         in_w     = (const float*)d_in[3];   // (4096,1024)
    const float*         conv_w   = (const float*)d_in[4];   // (2048,4)
    const float*         conv_b   = (const float*)d_in[5];
    const float*         xp_w     = (const float*)d_in[6];   // (96,2048)
    const float*         dt_w     = (const float*)d_in[7];   // (2048,64)
    const float*         dt_b     = (const float*)d_in[8];
    const float*         A_log    = (const float*)d_in[9];   // (2048,16)
    const float*         Dco      = (const float*)d_in[10];
    const float*         out_w    = (const float*)d_in[11];  // (1024,2048)
    float*               out      = (float*)d_out;

    // workspace carve-out (256B aligned)
    size_t off = 0;
    char* base = (char*)d_ws;
    auto alloc = [&](size_t bytes) -> void* {
        off = (off + 255) & ~(size_t)255;
        void* p = base + off;
        off += bytes;
        return p;
    };
    bf16_t* in_w_bf   = (bf16_t*)alloc((size_t)(2 * D_INNER) * D_MODEL * 2);
    bf16_t* xp_w_bf   = (bf16_t*)alloc((size_t)XDBL_LD * D_INNER * 2);  // padded
    bf16_t* dt_w_bf   = (bf16_t*)alloc((size_t)D_INNER * DT_RANK * 2);
    bf16_t* out_w_bf  = (bf16_t*)alloc((size_t)D_MODEL * D_INNER * 2);
    bf16_t* normed_bf = (bf16_t*)alloc((size_t)NTOK * D_MODEL * 2);
    float*  xz        = (float*) alloc((size_t)NTOK * 2 * D_INNER * 4);
    float*  x_c       = (float*) alloc((size_t)NTOK * D_INNER * 4);
    bf16_t* x_c_bf    = (bf16_t*)alloc((size_t)NTOK * D_INNER * 2);
    float*  x_dbl     = (float*) alloc((size_t)NTOK * XDBL_LD * 4);     // padded
    bf16_t* dtr_bf    = (bf16_t*)alloc((size_t)NTOK * DT_RANK * 2);
    float*  dt        = (float*) alloc((size_t)NTOK * D_INNER * 4);
    bf16_t* ybf       = (bf16_t*)alloc((size_t)NTOK * D_INNER * 2);

    auto cvt = [&](const float* src, bf16_t* dst, size_t n) {
        const int g = (int)((n + 255) / 256);
        f32_to_bf16_kernel<<<g, 256, 0, stream>>>(src, dst, n);
    };
    // 1. weights -> bf16 (x_proj zero-padded 96 -> 128 rows)
    cvt(in_w,  in_w_bf,  (size_t)(2 * D_INNER) * D_MODEL);
    {
        const size_t n = (size_t)XDBL_LD * D_INNER;
        f32_to_bf16_padrows_kernel<<<(int)((n + 255) / 256), 256, 0, stream>>>(
            xp_w, xp_w_bf, N_XDBL, D_INNER, n);
    }
    cvt(dt_w,  dt_w_bf,  (size_t)D_INNER * DT_RANK);
    cvt(out_w, out_w_bf, (size_t)D_MODEL * D_INNER);

    // 2. RMSNorm + residual
    rms_prep_kernel<<<NTOK, 256, 0, stream>>>(x, mask, rms_w, out, normed_bf);

    // 3. in_proj: xz[4096,4096] = normed * in_w^T
    wmma_gemm_kernel<<<dim3((2 * D_INNER) / 128, NTOK / 64), 256, 0, stream>>>(
        normed_bf, in_w_bf, xz, NTOK, 2 * D_INNER, D_MODEL, 0, nullptr, nullptr);

    // 4. segmented conv + SiLU
    {
        const size_t n = (size_t)NTOK * D_INNER;
        conv_silu_kernel<<<(int)((n + 255) / 256), 256, 0, stream>>>(
            xz, mask, conv_w, conv_b, x_c, x_c_bf, n);
    }

    // 5. x_proj: x_dbl[4096,128] = x_c * xp_w_pad^T (cols 96..127 are zero)
    wmma_gemm_kernel<<<dim3(XDBL_LD / 128, NTOK / 64), 256, 0, stream>>>(
        x_c_bf, xp_w_bf, x_dbl, NTOK, XDBL_LD, D_INNER, 0, nullptr, nullptr);

    // 6. dt_r slice -> bf16
    {
        const size_t n = (size_t)NTOK * DT_RANK;
        dtr_slice_kernel<<<(int)((n + 255) / 256), 256, 0, stream>>>(x_dbl, dtr_bf, n);
    }

    // 7. dt_proj: dt[4096,2048] = softplus(dt_r * dt_w^T + dt_b)
    wmma_gemm_kernel<<<dim3(D_INNER / 128, NTOK / 64), 256, 0, stream>>>(
        dtr_bf, dt_w_bf, dt, NTOK, D_INNER, DT_RANK, 1, dt_b, nullptr);

    // 8. SSM scan -> ybf
    scan_kernel<<<dim3(D_INNER / 256, BATCH), 256, 0, stream>>>(
        x_dbl, dt, x_c, xz, mask, A_log, Dco, ybf);

    // 9. out_proj: d_out += (ybf * out_w^T) * mask
    wmma_gemm_kernel<<<dim3(D_MODEL / 128, NTOK / 64), 256, 0, stream>>>(
        ybf, out_w_bf, out, NTOK, D_MODEL, D_INNER, 2, nullptr, mask);
}
